// NodeEdge_Layer_90975997264165
// MI455X (gfx1250) — matrix operand order repaired and checked by reading, not compiled, not run on hardware
//
#include <hip/hip_runtime.h>
#include <hip/hip_bf16.h>

#define N_NODES 25000
#define N_EDGES 400000
#define NODE_OUT_ELEMS (N_NODES * 128)

typedef __attribute__((ext_vector_type(16))) __bf16 v16bf;
typedef __attribute__((ext_vector_type(2)))  __bf16 v2bf;
typedef __attribute__((ext_vector_type(8)))  float  v8f;

union ABfrag {
    unsigned int u[8];
    uint4        q[2];
    v16bf        v;
};

// pack two f32 into one dword of bf16x2 — native v_cvt_pk_bf16_f32
__device__ __forceinline__ unsigned int pack2bf(float lo, float hi) {
#if __has_builtin(__builtin_amdgcn_cvt_pk_bf16_f32)
    union { v2bf v; unsigned int u; } cv;
    cv.v = __builtin_amdgcn_cvt_pk_bf16_f32(lo, hi);
    return cv.u;
#else
    union { v2bf v; unsigned int u; } cv;
    cv.v = (v2bf){(__bf16)lo, (__bf16)hi};
    return cv.u;
#endif
}

__device__ __forceinline__ v8f wmma_bf16(const ABfrag& a, const ABfrag& b, v8f c) {
    return __builtin_amdgcn_wmma_f32_16x16x32_bf16(false, a.v, false, b.v,
                                                   (short)0, c, false, false);
}

// Build an A fragment (16x32 bf16) from a row of f32 source data.
// Lane holds row m = lane%16; K pairs: v<4 -> k=2v+8*half, v>=4 -> k=16+2(v-4)+8*half.
// Pairs for v..v+1 are K-contiguous, so each half is two aligned float4 loads.
__device__ __forceinline__ void load_afrag(ABfrag& A, const float* __restrict__ src,
                                           int half) {
    const float* s0 = src + 8 * half;        // covers v=0..3  (k = 8h .. 8h+7)
    const float* s1 = src + 16 + 8 * half;   // covers v=4..7  (k = 16+8h .. 16+8h+7)
    float4 p0 = *(const float4*)(s0);
    float4 p1 = *(const float4*)(s0 + 4);
    float4 p2 = *(const float4*)(s1);
    float4 p3 = *(const float4*)(s1 + 4);
    A.u[0] = pack2bf(p0.x, p0.y); A.u[1] = pack2bf(p0.z, p0.w);
    A.u[2] = pack2bf(p1.x, p1.y); A.u[3] = pack2bf(p1.z, p1.w);
    A.u[4] = pack2bf(p2.x, p2.y); A.u[5] = pack2bf(p2.z, p2.w);
    A.u[6] = pack2bf(p3.x, p3.y); A.u[7] = pack2bf(p3.z, p3.w);
}

// Load a pre-swizzled B fragment from LDS: lane-contiguous 32B -> 2x ds_load_b128.
__device__ __forceinline__ void load_bfrag(ABfrag& B, const unsigned int* wf) {
    const uint4* q = (const uint4*)wf;
    B.q[0] = q[0];
    B.q[1] = q[1];
}

// Convert a row-major f32 weight matrix W[K x 128] into bf16 B-fragments in LDS.
// Fragment f = kchunk*8 + ntile; per fragment: 32 lanes x 8 dwords, lane-contiguous.
// B layout (16-bit, 32x16): lane holds column n = nt*16 + lane%16;
// K pairs: k = kchunk*32 + (v/4)*16 + (lane/16)*8 + 2(v%4) (+0/1).
__device__ __forceinline__ void build_wfrags(const float* __restrict__ W, int K,
                                             unsigned int* dst, int tid, int nthreads) {
    const int total = (K / 32) * 8 * 256;
    for (int d = tid; d < total; d += nthreads) {
        int f    = d >> 8;
        int rem  = d & 255;
        int lane = rem >> 3;
        int v    = rem & 7;
        int c    = f >> 3, nt = f & 7;
        int n    = nt * 16 + (lane & 15);
        int k    = c * 32 + (v & 4) * 4 + (lane >> 4) * 8 + 2 * (v & 3);
        dst[d] = pack2bf(W[k * 128 + n], W[(k + 1) * 128 + n]);
    }
}

__global__ void zero_kernel(float4* __restrict__ p, int n4) {
    const float4 z = {0.0f, 0.0f, 0.0f, 0.0f};
    for (int i = blockIdx.x * blockDim.x + threadIdx.x; i < n4; i += gridDim.x * blockDim.x)
        p[i] = z;
}

// ---------------- Edge kernel: gather -> MLP(320->128->128) -> store + scatter-add ----
__global__ void __launch_bounds__(128, 1)
edge_kernel(const float* __restrict__ nf, const float* __restrict__ ea,
            const int* __restrict__ ei,
            const float* __restrict__ W1, const float* __restrict__ b1,
            const float* __restrict__ W2, const float* __restrict__ b2,
            float* __restrict__ eoutBase, float* __restrict__ agg) {
    extern __shared__ char smem[];
    unsigned int*  sW1 = (unsigned int*)smem;               // 80 frags * 256 dw = 80 KB
    unsigned int*  sW2 = sW1 + 80 * 256;                    // 32 frags        = 32 KB
    float*         sB1 = (float*)(sW2 + 32 * 256);          // 128 f32
    float*         sB2 = sB1 + 128;                         // 128 f32
    __bf16*        sH  = (__bf16*)(sB2 + 128);              // 4 waves * 16x128 bf16

    const int tid = threadIdx.x;
    build_wfrags(W1, 320, sW1, tid, 128);
    build_wfrags(W2, 128, sW2, tid, 128);
    sB1[tid] = b1[tid];
    sB2[tid] = b2[tid];
    __syncthreads();

    const int lane = tid & 31;
    const int m    = lane & 15;
    const int half = lane >> 4;
    const int wave = tid >> 5;
    __bf16* hbuf = sH + wave * 2048;

    const int gwave  = blockIdx.x * 4 + wave;
    const int nwaves = gridDim.x * 4;
    const int ntiles = N_EDGES / 16;

    const int* ei0 = ei;
    const int* ei1 = ei + N_EDGES;

    for (int t = gwave; t < ntiles; t += nwaves) {
        const int e = t * 16 + m;
        const float* srcA = nf + (size_t)ei0[e] * 128;
        const float* srcB = nf + (size_t)ei1[e] * 128;
        const float* srcC = ea + (size_t)e * 64;

        // accumulate from zero (SRC2 can fold to inline 0); bias added in epilogue
        v8f acc[8];
#pragma unroll
        for (int nt = 0; nt < 8; ++nt) acc[nt] = (v8f){};

        // GEMM1: [16x320] x [320x128], K chunks of 32
#pragma unroll
        for (int c = 0; c < 10; ++c) {
            const float* src = (c < 4) ? (srcA + c * 32)
                             : (c < 8) ? (srcB + (c - 4) * 32)
                                       : (srcC + (c - 8) * 32);
            ABfrag A;
            load_afrag(A, src, half);
            const unsigned int* wf = sW1 + c * 8 * 256 + lane * 8;
#pragma unroll
            for (int nt = 0; nt < 8; ++nt) {
                ABfrag B;
                load_bfrag(B, wf + nt * 256);
                acc[nt] = wmma_bf16(A, B, acc[nt]);
            }
        }

        // +bias, ReLU, bounce h through LDS as bf16 row-major [16x128]
#pragma unroll
        for (int nt = 0; nt < 8; ++nt) {
            float bv = sB1[nt * 16 + m];
#pragma unroll
            for (int r = 0; r < 8; ++r) {
                float x = fmaxf(acc[nt][r] + bv, 0.0f);
                hbuf[(r + 8 * half) * 128 + nt * 16 + m] = (__bf16)x;
            }
        }

        // GEMM2: [16x128] x [128x128]
        v8f acc2[8];
#pragma unroll
        for (int nt = 0; nt < 8; ++nt) acc2[nt] = (v8f){};
#pragma unroll
        for (int c = 0; c < 4; ++c) {
            ABfrag A;
            // A pairs are K-contiguous: two aligned 16B LDS reads per fragment
            A.q[0] = *(const uint4*)(hbuf + m * 128 + c * 32 + 8 * half);
            A.q[1] = *(const uint4*)(hbuf + m * 128 + c * 32 + 16 + 8 * half);
            const unsigned int* wf = sW2 + c * 8 * 256 + lane * 8;
#pragma unroll
            for (int nt = 0; nt < 8; ++nt) {
                ABfrag B;
                load_bfrag(B, wf + nt * 256);
                acc2[nt] = wmma_bf16(A, B, acc2[nt]);
            }
        }

        // +bias, store edge_out + scatter-add into agg (f32 HW atomics, L2-resident)
        float* eout = eoutBase + (size_t)t * 16 * 128;
#pragma unroll
        for (int r = 0; r < 8; ++r) {
            int row = r + 8 * half;
            int dn  = ei0[t * 16 + row];
            float* aggRow = agg + (size_t)dn * 128;
#pragma unroll
            for (int nt = 0; nt < 8; ++nt) {
                int col  = nt * 16 + m;
                float val = acc2[nt][r] + sB2[col];
                eout[row * 128 + col] = val;
                unsafeAtomicAdd(aggRow + col, val);
            }
        }
    }
}

// ---------------- Node kernel: concat(node_feats, agg) -> MLP(256->128->128) ----------
__global__ void __launch_bounds__(128, 1)
node_kernel(const float* __restrict__ nf, const float* __restrict__ agg,
            const float* __restrict__ W1, const float* __restrict__ b1,
            const float* __restrict__ W2, const float* __restrict__ b2,
            float* __restrict__ nout) {
    extern __shared__ char smem[];
    unsigned int*  sW1 = (unsigned int*)smem;               // 64 frags = 64 KB
    unsigned int*  sW2 = sW1 + 64 * 256;                    // 32 frags = 32 KB
    float*         sB1 = (float*)(sW2 + 32 * 256);
    float*         sB2 = sB1 + 128;
    __bf16*        sH  = (__bf16*)(sB2 + 128);

    const int tid = threadIdx.x;
    build_wfrags(W1, 256, sW1, tid, 128);
    build_wfrags(W2, 128, sW2, tid, 128);
    sB1[tid] = b1[tid];
    sB2[tid] = b2[tid];
    __syncthreads();

    const int lane = tid & 31;
    const int m    = lane & 15;
    const int half = lane >> 4;
    const int wave = tid >> 5;
    __bf16* hbuf = sH + wave * 2048;

    const int gwave  = blockIdx.x * 4 + wave;
    const int nwaves = gridDim.x * 4;
    const int ntiles = (N_NODES + 15) / 16;

    for (int t = gwave; t < ntiles; t += nwaves) {
        int n  = t * 16 + m;
        int ns = (n < N_NODES) ? n : 0;
        const float* srcA = nf  + (size_t)ns * 128;
        const float* srcB = agg + (size_t)ns * 128;

        v8f acc[8];
#pragma unroll
        for (int nt = 0; nt < 8; ++nt) acc[nt] = (v8f){};

        // GEMM1: [16x256] x [256x128]
#pragma unroll
        for (int c = 0; c < 8; ++c) {
            const float* src = (c < 4) ? (srcA + c * 32) : (srcB + (c - 4) * 32);
            ABfrag A;
            load_afrag(A, src, half);
            const unsigned int* wf = sW1 + c * 8 * 256 + lane * 8;
#pragma unroll
            for (int nt = 0; nt < 8; ++nt) {
                ABfrag B;
                load_bfrag(B, wf + nt * 256);
                acc[nt] = wmma_bf16(A, B, acc[nt]);
            }
        }

#pragma unroll
        for (int nt = 0; nt < 8; ++nt) {
            float bv = sB1[nt * 16 + m];
#pragma unroll
            for (int r = 0; r < 8; ++r) {
                float x = fmaxf(acc[nt][r] + bv, 0.0f);
                hbuf[(r + 8 * half) * 128 + nt * 16 + m] = (__bf16)x;
            }
        }

        // GEMM2: [16x128] x [128x128]
        v8f acc2[8];
#pragma unroll
        for (int nt = 0; nt < 8; ++nt) acc2[nt] = (v8f){};
#pragma unroll
        for (int c = 0; c < 4; ++c) {
            ABfrag A;
            A.q[0] = *(const uint4*)(hbuf + m * 128 + c * 32 + 8 * half);
            A.q[1] = *(const uint4*)(hbuf + m * 128 + c * 32 + 16 + 8 * half);
            const unsigned int* wf = sW2 + c * 8 * 256 + lane * 8;
#pragma unroll
            for (int nt = 0; nt < 8; ++nt) {
                ABfrag B;
                load_bfrag(B, wf + nt * 256);
                acc2[nt] = wmma_bf16(A, B, acc2[nt]);
            }
        }

#pragma unroll
        for (int r = 0; r < 8; ++r) {
            int row  = r + 8 * half;
            int nrow = t * 16 + row;
            if (nrow < N_NODES) {
#pragma unroll
                for (int nt = 0; nt < 8; ++nt) {
                    int col = nt * 16 + m;
                    nout[(size_t)nrow * 128 + col] = acc2[nt][r] + sB2[col];
                }
            }
        }
    }
}

extern "C" void kernel_launch(void* const* d_in, const int* in_sizes, int n_in,
                              void* d_out, int out_size, void* d_ws, size_t ws_size,
                              hipStream_t stream) {
    (void)in_sizes; (void)n_in; (void)out_size; (void)ws_size;

    const float* nf  = (const float*)d_in[0];
    const float* ea  = (const float*)d_in[1];
    const int*   ei  = (const int*)  d_in[2];
    const float* eW1 = (const float*)d_in[3];
    const float* eb1 = (const float*)d_in[4];
    const float* eW2 = (const float*)d_in[5];
    const float* eb2 = (const float*)d_in[6];
    const float* nW1 = (const float*)d_in[7];
    const float* nb1 = (const float*)d_in[8];
    const float* nW2 = (const float*)d_in[9];
    const float* nb2 = (const float*)d_in[10];

    float* nodeOut = (float*)d_out;                    // [25000 x 128]
    float* edgeOut = nodeOut + NODE_OUT_ELEMS;         // [400000 x 128]
    float* agg     = (float*)d_ws;                     // [25000 x 128] scratch

    zero_kernel<<<dim3(100), dim3(256), 0, stream>>>((float4*)agg, N_NODES * 128 / 4);

    const size_t eshm = (size_t)(80 + 32) * 256 * 4 + 256 * 4 + 4 * 2048 * 2; // 132096 B
    edge_kernel<<<dim3(512), dim3(128), eshm, stream>>>(nf, ea, ei, eW1, eb1, eW2, eb2,
                                                        edgeOut, agg);

    const size_t nshm = (size_t)(64 + 32) * 256 * 4 + 256 * 4 + 4 * 2048 * 2; // 115712 B
    node_kernel<<<dim3(160), dim3(128), nshm, stream>>>(nf, agg, nW1, nb1, nW2, nb2,
                                                        nodeOut);
}
